// VeRA_Linear_61813169324212
// MI455X (gfx1250) — matrix operand order repaired and checked by reading, not compiled, not run on hardware
//
#include <hip/hip_runtime.h>
#include <hip/hip_bf16.h>

// VeRA linear:  y = x @ (W + 4*diag(b) B diag(d) A)^T + bias
// Factored:     t = (x @ A^T) * (4*d)                (8192 x 256, bf16, in d_ws)
//               y = x @ W^T + t @ (b o B)^T + bias   (fused two-phase K loop)
// Main GEMM on v_wmma_f32_16x16x32_bf16 (f32 accumulate), double-buffered LDS,
// software-pipelined global->LDS staging (one barrier per K-step).

typedef __bf16 bf16_t;
typedef __attribute__((ext_vector_type(16))) __bf16 bf16x16;
typedef __attribute__((ext_vector_type(8)))  __bf16 bf16x8;
typedef __attribute__((ext_vector_type(8)))  float  f32x8;
typedef __attribute__((ext_vector_type(4)))  float  f32x4;

constexpr int   BM = 128;   // rows of x per workgroup
constexpr int   BN = 128;   // rows of W (output cols) per workgroup
constexpr int   BK = 32;    // one bf16 WMMA K step
constexpr int   LDT = 40;   // LDS pitch in bf16: 80B = 5*16B -> b128-aligned, conflict-free
constexpr float VERA_SCALE = 4.0f;

// ---- A-matrix fragment (16-bit 16x32 layout, ISA 7.12.2):
// lane: M = lane&15, half = lane>>4
// elem j in [0,8):  K = half*8 + j          (VGPR 0..3)
// elem j in [8,16): K = 16 + half*8 + (j-8) (VGPR 4..7)
__device__ __forceinline__ bf16x16 lds_fragA(const bf16_t* rowbase, int half) {
  bf16x8 lo = *reinterpret_cast<const bf16x8*>(rowbase + half * 8);
  bf16x8 hi = *reinterpret_cast<const bf16x8*>(rowbase + 16 + half * 8);
  bf16x16 r;
#pragma unroll
  for (int j = 0; j < 8; ++j) { r[j] = lo[j]; r[j + 8] = hi[j]; }
  return r;
}

// ---- B-matrix fragment (32x16): lane holds column N = lane&15;
// same K order as the A fragment:
// elem j in [0,8): K = half*8 + j; elem j in [8,16): K = 16 + half*8 + (j-8)
__device__ __forceinline__ bf16x16 lds_fragB(const bf16_t* rowbase, int half) {
  bf16x8 lo = *reinterpret_cast<const bf16x8*>(rowbase + half * 8);
  bf16x8 hi = *reinterpret_cast<const bf16x8*>(rowbase + 16 + half * 8);
  bf16x16 r;
#pragma unroll
  for (int j = 0; j < 8; ++j) { r[j] = lo[j]; r[j + 8] = hi[j]; }
  return r;
}

// 16-float strip held in registers while WMMAs of the previous tile run.
struct StripF32 { f32x4 v[4]; };

__device__ __forceinline__ StripF32 load_strip(const float* g) {
  StripF32 s;
  s.v[0] = *(const f32x4*)(g + 0);
  s.v[1] = *(const f32x4*)(g + 4);
  s.v[2] = *(const f32x4*)(g + 8);
  s.v[3] = *(const f32x4*)(g + 12);
  return s;
}

__device__ __forceinline__ void cvt_store_strip(bf16_t* smem, const StripF32& t, float scl) {
  bf16x8 p0, p1;
#pragma unroll
  for (int j = 0; j < 4; ++j) {
    p0[j]     = (bf16_t)(t.v[0][j] * scl);
    p0[j + 4] = (bf16_t)(t.v[1][j] * scl);
    p1[j]     = (bf16_t)(t.v[2][j] * scl);
    p1[j + 4] = (bf16_t)(t.v[3][j] * scl);
  }
  *reinterpret_cast<bf16x8*>(smem)     = p0;
  *reinterpret_cast<bf16x8*>(smem + 8) = p1;
}

// 8 WMMAs per wave per K-step: 4 M-tiles x 2 N-tiles of 16x16.
__device__ __forceinline__ void mma_step(const bf16_t* As, const bf16_t* Bs,
                                         int wm, int wn, int lrw, int half,
                                         f32x8 acc[4][2]) {
  bf16x16 af[4];
  bf16x16 bfrag[2];
#pragma unroll
  for (int mt = 0; mt < 4; ++mt)
    af[mt] = lds_fragA(As + (wm * 64 + mt * 16 + lrw) * LDT, half);
#pragma unroll
  for (int nt = 0; nt < 2; ++nt)
    bfrag[nt] = lds_fragB(Bs + (wn * 32 + nt * 16 + lrw) * LDT, half);
#pragma unroll
  for (int mt = 0; mt < 4; ++mt)
#pragma unroll
    for (int nt = 0; nt < 2; ++nt)
      acc[mt][nt] = __builtin_amdgcn_wmma_f32_16x16x32_bf16(
          false, af[mt], false, bfrag[nt], (short)0, acc[mt][nt], false, false);
}

// Kernel 1: t[T][R] = bf16( (x @ A^T) * 4*d[r] )
__global__ __launch_bounds__(256)
void vera_xa_kernel(const float* __restrict__ x, const float* __restrict__ A,
                    const float* __restrict__ dvec, bf16_t* __restrict__ txA,
                    int T, int IN, int R) {
  __shared__ alignas(16) bf16_t As[2][BM * LDT];
  __shared__ alignas(16) bf16_t Bs[2][BN * LDT];
  const int tid  = threadIdx.x;
  const int lane = tid & 31, wave = tid >> 5;
  const int wm = wave & 1, wn = wave >> 1;       // 2x4 wave grid
  const int lrw = lane & 15, half = lane >> 4;
  const int m0 = blockIdx.y * BM, n0 = blockIdx.x * BN;
  const int lrow = tid >> 1, lcol = (tid & 1) * 16;
  const int soff = lrow * LDT + lcol;

  f32x8 acc[4][2];
  const f32x8 fzero = {0.f, 0.f, 0.f, 0.f, 0.f, 0.f, 0.f, 0.f};
#pragma unroll
  for (int mt = 0; mt < 4; ++mt)
#pragma unroll
    for (int nt = 0; nt < 2; ++nt) acc[mt][nt] = fzero;

  const float* gx = x + (size_t)(m0 + lrow) * IN + lcol;
  const float* ga = A + (size_t)(n0 + lrow) * IN + lcol;

  // prologue: stage tile 0 into buffer 0
  {
    StripF32 ra = load_strip(gx);
    StripF32 rb = load_strip(ga);
    cvt_store_strip(&As[0][soff], ra, 1.0f);
    cvt_store_strip(&Bs[0][soff], rb, 1.0f);
  }
  __syncthreads();

  int cur = 0;
  for (int k0 = 0; k0 < IN; k0 += BK) {
    const bool last = (k0 + BK >= IN);
    StripF32 ra, rb;
    if (!last) {                       // issue next tile's loads before the WMMAs
      ra = load_strip(gx + k0 + BK);
      rb = load_strip(ga + k0 + BK);
    }
    mma_step(As[cur], Bs[cur], wm, wn, lrw, half, acc);
    if (!last) {
      cvt_store_strip(&As[cur ^ 1][soff], ra, 1.0f);
      cvt_store_strip(&Bs[cur ^ 1][soff], rb, 1.0f);
      __syncthreads();                 // one barrier per K-step
      cur ^= 1;
    }
  }

#pragma unroll
  for (int mt = 0; mt < 4; ++mt)
#pragma unroll
    for (int nt = 0; nt < 2; ++nt) {
      const int col = n0 + wn * 32 + nt * 16 + lrw;
      const float ds = VERA_SCALE * dvec[col];
#pragma unroll
      for (int v = 0; v < 8; ++v) {
        const int row = m0 + wm * 64 + mt * 16 + half * 8 + v;
        txA[(size_t)row * R + col] = (bf16_t)(acc[mt][nt][v] * ds);
      }
    }
}

// Kernel 2: y = x @ W^T  (K = IN)  +  t @ (b o B)^T  (K = R, fused)  + bias
__global__ __launch_bounds__(256)
void vera_main_gemm(const float* __restrict__ x, const float* __restrict__ W,
                    const float* __restrict__ bias, const float* __restrict__ Bmat,
                    const float* __restrict__ bvec, const bf16_t* __restrict__ txA,
                    float* __restrict__ out, int T, int IN, int OUT, int R) {
  __shared__ alignas(16) bf16_t As[2][BM * LDT];
  __shared__ alignas(16) bf16_t Bs[2][BN * LDT];
  const int tid  = threadIdx.x;
  const int lane = tid & 31, wave = tid >> 5;
  const int wm = wave & 1, wn = wave >> 1;
  const int lrw = lane & 15, half = lane >> 4;
  const int m0 = blockIdx.y * BM, n0 = blockIdx.x * BN;
  const int lrow = tid >> 1, lcol = (tid & 1) * 16;
  const int soff = lrow * LDT + lcol;

  f32x8 acc[4][2];
  const f32x8 fzero = {0.f, 0.f, 0.f, 0.f, 0.f, 0.f, 0.f, 0.f};
#pragma unroll
  for (int mt = 0; mt < 4; ++mt)
#pragma unroll
    for (int nt = 0; nt < 2; ++nt) acc[mt][nt] = fzero;

  const float* gx = x + (size_t)(m0 + lrow) * IN + lcol;
  const float* gw = W + (size_t)(n0 + lrow) * IN + lcol;

  // ---- Phase 1: dense base-weight GEMM, K = IN (pipelined, double-buffered) ----
  {
    StripF32 ra = load_strip(gx);
    StripF32 rb = load_strip(gw);
    cvt_store_strip(&As[0][soff], ra, 1.0f);
    cvt_store_strip(&Bs[0][soff], rb, 1.0f);
  }
  __syncthreads();

  int cur = 0;
  for (int k0 = 0; k0 < IN; k0 += BK) {
    const bool last = (k0 + BK >= IN);
    StripF32 ra, rb;
    if (!last) {
      ra = load_strip(gx + k0 + BK);
      rb = load_strip(gw + k0 + BK);
    }
    mma_step(As[cur], Bs[cur], wm, wn, lrw, half, acc);
    if (!last) {
      cvt_store_strip(&As[cur ^ 1][soff], ra, 1.0f);
      cvt_store_strip(&Bs[cur ^ 1][soff], rb, 1.0f);
      __syncthreads();
      cur ^= 1;
    }
  }
  __syncthreads();   // all reads of phase-1 tiles complete before buffers are reused

  // ---- Phase 2: fused low-rank correction, K = R (t is pre-scaled by 4*d) ----
  const float   bscl = bvec[n0 + lrow];                       // folds diag(b) into B tile
  const bf16_t* gt   = txA + (size_t)(m0 + lrow) * R + lcol;
  const float*  gb   = Bmat + (size_t)(n0 + lrow) * R + lcol;

  {
    bf16x8 u0 = *reinterpret_cast<const bf16x8*>(gt);
    bf16x8 u1 = *reinterpret_cast<const bf16x8*>(gt + 8);
    StripF32 rb = load_strip(gb);
    bf16_t* s = &As[0][soff];
    *reinterpret_cast<bf16x8*>(s)     = u0;
    *reinterpret_cast<bf16x8*>(s + 8) = u1;
    cvt_store_strip(&Bs[0][soff], rb, bscl);
  }
  __syncthreads();

  cur = 0;
  for (int r0 = 0; r0 < R; r0 += BK) {
    const bool last = (r0 + BK >= R);
    bf16x8 u0, u1;
    StripF32 rb;
    if (!last) {
      u0 = *reinterpret_cast<const bf16x8*>(gt + r0 + BK);
      u1 = *reinterpret_cast<const bf16x8*>(gt + r0 + BK + 8);
      rb = load_strip(gb + r0 + BK);
    }
    mma_step(As[cur], Bs[cur], wm, wn, lrw, half, acc);
    if (!last) {
      bf16_t* s = &As[cur ^ 1][soff];
      *reinterpret_cast<bf16x8*>(s)     = u0;
      *reinterpret_cast<bf16x8*>(s + 8) = u1;
      cvt_store_strip(&Bs[cur ^ 1][soff], rb, bscl);
      __syncthreads();
      cur ^= 1;
    }
  }

  // ---- Epilogue: C/D layout (ISA 7.12.2): M = half*8 + v, N = lane&15 ----
#pragma unroll
  for (int mt = 0; mt < 4; ++mt)
#pragma unroll
    for (int nt = 0; nt < 2; ++nt) {
      const int col = n0 + wn * 32 + nt * 16 + lrw;
      const float bv = bias[col];
#pragma unroll
      for (int v = 0; v < 8; ++v) {
        const int row = m0 + wm * 64 + mt * 16 + half * 8 + v;
        out[(size_t)row * OUT + col] = acc[mt][nt][v] + bv;
      }
    }
}

extern "C" void kernel_launch(void* const* d_in, const int* in_sizes, int n_in,
                              void* d_out, int out_size, void* d_ws, size_t ws_size,
                              hipStream_t stream) {
  const float* x    = (const float*)d_in[0];   // (T, IN)
  const float* W    = (const float*)d_in[1];   // (OUT, IN)
  const float* bias = (const float*)d_in[2];   // (OUT)
  const float* A    = (const float*)d_in[3];   // (R, IN)
  const float* Bm   = (const float*)d_in[4];   // (OUT, R)
  const float* dvec = (const float*)d_in[5];   // (R)
  const float* bvec = (const float*)d_in[6];   // (OUT)
  float* out = (float*)d_out;

  const int R   = in_sizes[5];
  const int OUT = in_sizes[2];
  const int IN  = in_sizes[3] / R;
  const int T   = in_sizes[0] / IN;

  bf16_t* txA = (bf16_t*)d_ws;   // T*R bf16 = 4 MB

  dim3 blk(256);
  vera_xa_kernel<<<dim3(R / BN, T / BM), blk, 0, stream>>>(x, A, dvec, txA, T, IN, R);
  vera_main_gemm<<<dim3(OUT / BN, T / BM), blk, 0, stream>>>(x, W, bias, Bm, bvec, txA, out,
                                                             T, IN, OUT, R);
}
